// Graphormer3D_65944927862854
// MI455X (gfx1250) — matrix operand order repaired and checked by reading, not compiled, not run on hardware
//
#include <hip/hip_runtime.h>
#include <cmath>

// ---------------- types ----------------
typedef __attribute__((ext_vector_type(16))) __bf16 v16bf;
typedef __attribute__((ext_vector_type(8)))  float  v8f;

union FragB { v16bf v; uint4 u[2]; };

static __device__ __forceinline__ v8f zero8() {
  v8f z = {0.f,0.f,0.f,0.f,0.f,0.f,0.f,0.f};
  return z;
}

static __device__ __forceinline__ unsigned short f2bf(float x) {
  unsigned int u = __float_as_uint(x);
  unsigned int r = u + 0x7FFFu + ((u >> 16) & 1u);
  return (unsigned short)(r >> 16);
}

static __device__ __forceinline__ unsigned int pk2(float a, float b) {
#if __has_builtin(__builtin_amdgcn_cvt_pk_bf16_f32)
  auto r = __builtin_amdgcn_cvt_pk_bf16_f32(a, b);
  unsigned int u;
  __builtin_memcpy(&u, &r, 4);
  return u;
#else
  return (unsigned int)f2bf(a) | ((unsigned int)f2bf(b) << 16);
#endif
}

// async global->LDS copy of 32 bytes (two B128 async ops), tracked by ASYNCcnt
static __device__ __forceinline__ void async_copy32(unsigned lds_off, const void* sbase, unsigned goff) {
  asm volatile("global_load_async_to_lds_b128 %0, %1, %2\n\t"
               "global_load_async_to_lds_b128 %0, %1, %2 offset:16"
               :: "v"(lds_off), "v"(goff), "s"(sbase) : "memory");
}
static __device__ __forceinline__ void wait_async0() {
  asm volatile("s_wait_asynccnt 0x0" ::: "memory");
}

#define WMMA_BF16(A,Bv,C) __builtin_amdgcn_wmma_f32_16x16x32_bf16(false,(A),false,(Bv),(short)0,(C),false,false)

// ---------------- dims ----------------
#define BB 4
#define NN 256
#define DD 768
#define HH 32
#define LL 12
#define FF 3072
#define KK 128
#define HD 24
#define BHN (BB*HH)

// ================= weight f32 -> bf16 =================
__global__ void f2bf_k(const float* __restrict__ src, unsigned short* __restrict__ dst, int n) {
  int i = blockIdx.x * 256 + threadIdx.x;
  if (i < n) dst[i] = f2bf(src[i]);
}

// ================= generic GEMM: out[M,Nout] = X[M,K] @ Wb[Nout,K]^T (+bias)(gelu)(+resid) ======
template<bool GELU, bool RESID>
__global__ void __launch_bounds__(256) gemm_bf16_k(
    const float* __restrict__ X, int ldx,
    const unsigned short* __restrict__ Wb,
    const float* __restrict__ bias,
    const float* __restrict__ resid, int ldr,
    float* __restrict__ out, int ldo,
    int K)
{
  __shared__ unsigned short As[128 * 40];
  __shared__ unsigned short Bs[128 * 40];
  const int m0 = blockIdx.x * 128;
  const int n0 = blockIdx.y * 128;
  const int t  = threadIdx.x;
  const int w  = t >> 5;
  const int l  = t & 31;
  const int g  = l >> 4;
  const int ln = l & 15;
  const int arow = t >> 1;
  const int acol = (t & 1) * 16;

  const unsigned lds_b = (unsigned)(uintptr_t)(const void*)(Bs + arow * 40 + acol);

  v8f acc[8];
#pragma unroll
  for (int i = 0; i < 8; i++) acc[i] = zero8();

  for (int k0 = 0; k0 < K; k0 += 32) {
    // async stage B tile (bf16 weights, global -> LDS, no VGPR round trip)
    {
      unsigned goff = (unsigned)(((size_t)(n0 + arow) * K + k0 + acol) * sizeof(unsigned short));
      async_copy32(lds_b, (const void*)Wb, goff);
    }
    // stage A tile: f32 -> bf16 in VALU, ds_store
    {
      const float* srow = X + (size_t)(m0 + arow) * ldx + k0 + acol;
      const float4* s4 = (const float4*)srow;
      float4 f0 = s4[0], f1 = s4[1], f2 = s4[2], f3 = s4[3];
      unsigned int* d32 = (unsigned int*)(As + arow * 40 + acol);
      d32[0] = pk2(f0.x, f0.y); d32[1] = pk2(f0.z, f0.w);
      d32[2] = pk2(f1.x, f1.y); d32[3] = pk2(f1.z, f1.w);
      d32[4] = pk2(f2.x, f2.y); d32[5] = pk2(f2.z, f2.w);
      d32[6] = pk2(f3.x, f3.y); d32[7] = pk2(f3.z, f3.w);
      if (k0 + 32 < K) __builtin_prefetch(srow + 32, 0, 0);
    }
    wait_async0();
    __syncthreads();
    FragB a;
    {
      const unsigned short* ap = As + (w * 16 + ln) * 40;
      a.u[0] = *(const uint4*)(ap + 16 * g);
      a.u[1] = *(const uint4*)(ap + 16 * g + 8);
    }
    FragB bf[8];
#pragma unroll
    for (int ct = 0; ct < 8; ct++) {
      const unsigned short* bp = Bs + (ct * 16 + ln) * 40 + 16 * g;
      bf[ct].u[0] = *(const uint4*)(bp);
      bf[ct].u[1] = *(const uint4*)(bp + 8);
    }
#pragma unroll
    for (int ct = 0; ct < 8; ct++) acc[ct] = WMMA_BF16(a.v, bf[ct].v, acc[ct]);
    __syncthreads();
  }
#pragma unroll
  for (int ct = 0; ct < 8; ct++) {
    int col = n0 + ct * 16 + ln;
    float bv = bias[col];
#pragma unroll
    for (int r = 0; r < 8; r++) {
      int row = m0 + w * 16 + r + 8 * g;
      float v = acc[ct][r] + bv;
      if (GELU) v = 0.5f * v * (1.0f + erff(v * 0.70710678118f));
      if (RESID) v += resid[(size_t)row * ldr + col];
      out[(size_t)row * ldo + col] = v;
    }
  }
}

// ================= GBF features =================
__global__ void __launch_bounds__(128) gbf_k(
    const int* __restrict__ atoms, const float* __restrict__ pos,
    const float* __restrict__ means, const float* __restrict__ stds,
    const float* __restrict__ mulT, const float* __restrict__ addT,
    unsigned short* __restrict__ gbf, float* __restrict__ edge_sum)
{
  __shared__ float xl[NN];
  __shared__ int padl[NN];
  int blk = blockIdx.x;
  int b = blk >> 8, i = blk & 255;
  int t = threadIdx.x;
  float pix = pos[(b * NN + i) * 3 + 0];
  float piy = pos[(b * NN + i) * 3 + 1];
  float piz = pos[(b * NN + i) * 3 + 2];
  int ai = atoms[b * NN + i];
  for (int j = t; j < NN; j += 128) {
    float dx = pos[(b * NN + j) * 3 + 0] - pix;
    float dy = pos[(b * NN + j) * 3 + 1] - piy;
    float dz = pos[(b * NN + j) * 3 + 2] - piz;
    float dist = sqrtf(fmaxf(dx * dx + dy * dy + dz * dz, 1e-24f));
    int aj = atoms[b * NN + j];
    int et = ai * 64 + aj;
    xl[j] = mulT[et] * dist + addT[et];
    padl[j] = (aj == 0);
  }
  __syncthreads();
  int k = t;
  float mean = means[k];
  float sd = fabsf(stds[k]) + 1e-5f;
  float inv = 1.0f / sd;
  float coef = 0.3989422804014327f * inv;
  float sum = 0.f;
  size_t base = ((size_t)(b * NN + i)) * NN * KK;
  for (int j = 0; j < NN; j++) {
    float z = (xl[j] - mean) * inv;
    float gv = __expf(-0.5f * z * z) * coef;
    gbf[base + (size_t)j * KK + k] = f2bf(gv);
    if (!padl[j]) sum += gv;
  }
  edge_sum[(size_t)(b * NN + i) * KK + k] = sum;
}

// ================= fused bias MLP: gelu(gbf@W1^T+b1)@W2^T+b2 -> bias[B,H,N,N] =================
__global__ void __launch_bounds__(256) bias_mlp_k(
    const unsigned short* __restrict__ gbf,
    const unsigned short* __restrict__ w1b, const float* __restrict__ b1,
    const unsigned short* __restrict__ w2b, const float* __restrict__ b2,
    const int* __restrict__ atoms, float* __restrict__ biasout)
{
  __shared__ unsigned short Ts[128 * 136];
  const int m0 = blockIdx.x * 128;
  const int t = threadIdx.x;
  const int w = t >> 5, l = t & 31, g = l >> 4, ln = l & 15;

  v8f acc[8];
#pragma unroll
  for (int i = 0; i < 8; i++) acc[i] = zero8();
#pragma unroll
  for (int ks = 0; ks < 4; ks++) {
    int k0 = ks * 32;
    FragB a;
    const unsigned short* ap = gbf + (size_t)(m0 + w * 16 + ln) * KK + k0;
    a.u[0] = *(const uint4*)(ap + 16 * g);
    a.u[1] = *(const uint4*)(ap + 16 * g + 8);
    FragB bf[8];
#pragma unroll
    for (int ct = 0; ct < 8; ct++) {
      const unsigned short* bp = w1b + (ct * 16 + ln) * KK + k0 + 16 * g;
      bf[ct].u[0] = *(const uint4*)(bp);
      bf[ct].u[1] = *(const uint4*)(bp + 8);
    }
#pragma unroll
    for (int ct = 0; ct < 8; ct++) acc[ct] = WMMA_BF16(a.v, bf[ct].v, acc[ct]);
  }
#pragma unroll
  for (int ct = 0; ct < 8; ct++) {
    float bv = b1[ct * 16 + ln];
#pragma unroll
    for (int r = 0; r < 8; r++) {
      float v = acc[ct][r] + bv;
      v = 0.5f * v * (1.0f + erff(v * 0.70710678118f));
      Ts[(w * 16 + r + 8 * g) * 136 + ct * 16 + ln] = f2bf(v);
    }
  }
  __syncthreads();
  v8f a2[2]; a2[0] = zero8(); a2[1] = zero8();
#pragma unroll
  for (int ks = 0; ks < 4; ks++) {
    int k0 = ks * 32;
    FragB a;
    const unsigned short* ap = Ts + (w * 16 + ln) * 136 + k0;
    a.u[0] = *(const uint4*)(ap + 16 * g);
    a.u[1] = *(const uint4*)(ap + 16 * g + 8);
#pragma unroll
    for (int ct = 0; ct < 2; ct++) {
      FragB bf;
      const unsigned short* bp = w2b + (ct * 16 + ln) * KK + k0 + 16 * g;
      bf.u[0] = *(const uint4*)(bp);
      bf.u[1] = *(const uint4*)(bp + 8);
      a2[ct] = WMMA_BF16(a.v, bf.v, a2[ct]);
    }
  }
#pragma unroll
  for (int ct = 0; ct < 2; ct++) {
    int h = ct * 16 + ln;
    float bv = b2[h];
#pragma unroll
    for (int r = 0; r < 8; r++) {
      int row = m0 + w * 16 + r + 8 * g;
      int b = row >> 16, i = (row >> 8) & 255, j = row & 255;
      float v = a2[ct][r] + bv;
      if (atoms[b * NN + j] == 0) v = -INFINITY;
      biasout[(((size_t)(b * HH + h) * NN + i) * NN) + j] = v;
    }
  }
}

// ================= misc small kernels =================
__global__ void node_emb_k(const int* __restrict__ atoms, const int* __restrict__ tags,
                           const float* __restrict__ atom_emb, const float* __restrict__ tag_emb,
                           float* __restrict__ h)
{
  int row = blockIdx.x, t = threadIdx.x;
  int a = atoms[row], tg = tags[row];
  for (int c = t; c < DD; c += 256)
    h[(size_t)row * DD + c] = tag_emb[tg * DD + c] + atom_emb[a * DD + c];
}

__global__ void __launch_bounds__(256) ln_k(const float* __restrict__ x,
                                            const float* __restrict__ gamma,
                                            const float* __restrict__ beta,
                                            float* __restrict__ y)
{
  __shared__ float red[256];
  int row = blockIdx.x, t = threadIdx.x;
  const float* xr = x + (size_t)row * DD;
  float v0 = xr[t], v1 = xr[t + 256], v2 = xr[t + 512];
  red[t] = v0 + v1 + v2; __syncthreads();
  for (int o = 128; o > 0; o >>= 1) { if (t < o) red[t] += red[t + o]; __syncthreads(); }
  float mean = red[0] / (float)DD; __syncthreads();
  float d0 = v0 - mean, d1 = v1 - mean, d2 = v2 - mean;
  red[t] = d0 * d0 + d1 * d1 + d2 * d2; __syncthreads();
  for (int o = 128; o > 0; o >>= 1) { if (t < o) red[t] += red[t + o]; __syncthreads(); }
  float rstd = rsqrtf(red[0] / (float)DD + 1e-5f);
  float* yr = y + (size_t)row * DD;
  yr[t]       = d0 * rstd * gamma[t]       + beta[t];
  yr[t + 256] = d1 * rstd * gamma[t + 256] + beta[t + 256];
  yr[t + 512] = d2 * rstd * gamma[t + 512] + beta[t + 512];
}

__global__ void pack_qkv_k(const float* __restrict__ qkv,
                           unsigned short* __restrict__ qb, unsigned short* __restrict__ kb,
                           unsigned short* __restrict__ vtb, float scale)
{
  int bh = blockIdx.x;
  int b = bh >> 5, hh = bh & 31;
  int j = threadIdx.x;
  const float* rowp = qkv + ((size_t)(b * NN + j)) * (3 * DD) + hh * HD;
  for (int d = 0; d < 32; d++) {
    float qv = (d < HD) ? rowp[d] * scale : 0.f;
    float kv = (d < HD) ? rowp[DD + d] : 0.f;
    float vv = (d < HD) ? rowp[2 * DD + d] : 0.f;
    size_t rb = ((size_t)bh * NN + j) * 32 + d;
    qb[rb] = f2bf(qv);
    kb[rb] = f2bf(kv);
    vtb[((size_t)bh * 32 + d) * NN + j] = f2bf(vv);
  }
}

// ================= attention (standard): o = softmax(QK^T + bias) V =================
__global__ void __launch_bounds__(128) attn_k(
    const unsigned short* __restrict__ qb, const unsigned short* __restrict__ kb,
    const unsigned short* __restrict__ vtb, const float* __restrict__ biasB,
    float* __restrict__ o)
{
  __shared__ unsigned short P[64 * 264];
  int bh = blockIdx.x;
  int qtr = blockIdx.y;
  int b = bh >> 5, hh = bh & 31;
  int i0 = qtr * 64;
  int t = threadIdx.x;
  int w = t >> 5, l = t & 31, g = l >> 4, ln = l & 15;

  FragB qf;
  {
    const unsigned short* qp = qb + ((size_t)bh * NN + i0 + w * 16 + ln) * 32;
    qf.u[0] = *(const uint4*)(qp + 16 * g);
    qf.u[1] = *(const uint4*)(qp + 16 * g + 8);
  }
  v8f sc[16];
#pragma unroll
  for (int ct = 0; ct < 16; ct++) {
    FragB kf;
    const unsigned short* kp = kb + ((size_t)bh * NN + ct * 16 + ln) * 32 + 16 * g;
    kf.u[0] = *(const uint4*)kp;
    kf.u[1] = *(const uint4*)(kp + 8);
    sc[ct] = WMMA_BF16(qf.v, kf.v, zero8());
  }
  const float* bp = biasB + (size_t)bh * NN * NN;
#pragma unroll
  for (int ct = 0; ct < 16; ct++) {
    int j = ct * 16 + ln;
#pragma unroll
    for (int r = 0; r < 8; r++) {
      int i = i0 + w * 16 + r + 8 * g;
      sc[ct][r] += bp[(size_t)i * NN + j];
    }
  }
#pragma unroll
  for (int r = 0; r < 8; r++) {
    float m = -INFINITY;
#pragma unroll
    for (int ct = 0; ct < 16; ct++) m = fmaxf(m, sc[ct][r]);
    for (int msk = 1; msk < 16; msk <<= 1) m = fmaxf(m, __shfl_xor(m, msk));
    float s = 0.f;
#pragma unroll
    for (int ct = 0; ct < 16; ct++) { float e = __expf(sc[ct][r] - m); sc[ct][r] = e; s += e; }
    for (int msk = 1; msk < 16; msk <<= 1) s += __shfl_xor(s, msk);
    float inv = 1.0f / s;
#pragma unroll
    for (int ct = 0; ct < 16; ct++) sc[ct][r] *= inv;
  }
#pragma unroll
  for (int ct = 0; ct < 16; ct++) {
    int j = ct * 16 + ln;
#pragma unroll
    for (int r = 0; r < 8; r++) {
      int ml = w * 16 + r + 8 * g;
      P[ml * 264 + j] = f2bf(sc[ct][r]);
    }
  }
  __syncthreads();
  v8f ov[2]; ov[0] = zero8(); ov[1] = zero8();
#pragma unroll
  for (int ks = 0; ks < 8; ks++) {
    int k0 = ks * 32;
    FragB pf;
    const unsigned short* pp = P + (w * 16 + ln) * 264 + k0;
    pf.u[0] = *(const uint4*)(pp + 16 * g);
    pf.u[1] = *(const uint4*)(pp + 16 * g + 8);
#pragma unroll
    for (int ct = 0; ct < 2; ct++) {
      FragB vf;
      const unsigned short* vp = vtb + ((size_t)bh * 32 + ct * 16 + ln) * NN + k0 + 16 * g;
      vf.u[0] = *(const uint4*)vp;
      vf.u[1] = *(const uint4*)(vp + 8);
      ov[ct] = WMMA_BF16(pf.v, vf.v, ov[ct]);
    }
  }
#pragma unroll
  for (int ct = 0; ct < 2; ct++) {
    int d = ct * 16 + ln;
    if (d < HD) {
#pragma unroll
      for (int r = 0; r < 8; r++) {
        int i = i0 + w * 16 + r + 8 * g;
        o[((size_t)(b * NN + i)) * DD + hh * HD + d] = ov[ct][r];
      }
    }
  }
}

// ================= node-head attention with delta weighting =================
__global__ void __launch_bounds__(128) attn_node_k(
    const unsigned short* __restrict__ qb, const unsigned short* __restrict__ kb,
    const unsigned short* __restrict__ vtb, const float* __restrict__ biasB,
    const float* __restrict__ pos,
    const float* __restrict__ f1w, const float* __restrict__ f2w, const float* __restrict__ f3w,
    float* __restrict__ outp)
{
  __shared__ unsigned short P[64 * 264];
  __shared__ float psx[NN], psy[NN], psz[NN];
  int bh = blockIdx.x;
  int qtr = blockIdx.y;
  int b = bh >> 5, hh = bh & 31;
  int i0 = qtr * 64;
  int t = threadIdx.x;
  int w = t >> 5, l = t & 31, g = l >> 4, ln = l & 15;

  for (int jj = t; jj < NN; jj += 128) {
    psx[jj] = pos[(b * NN + jj) * 3 + 0];
    psy[jj] = pos[(b * NN + jj) * 3 + 1];
    psz[jj] = pos[(b * NN + jj) * 3 + 2];
  }

  FragB qf;
  {
    const unsigned short* qp = qb + ((size_t)bh * NN + i0 + w * 16 + ln) * 32;
    qf.u[0] = *(const uint4*)(qp + 16 * g);
    qf.u[1] = *(const uint4*)(qp + 16 * g + 8);
  }
  v8f sc[16];
#pragma unroll
  for (int ct = 0; ct < 16; ct++) {
    FragB kf;
    const unsigned short* kp = kb + ((size_t)bh * NN + ct * 16 + ln) * 32 + 16 * g;
    kf.u[0] = *(const uint4*)kp;
    kf.u[1] = *(const uint4*)(kp + 8);
    sc[ct] = WMMA_BF16(qf.v, kf.v, zero8());
  }
  const float* bp = biasB + (size_t)bh * NN * NN;
#pragma unroll
  for (int ct = 0; ct < 16; ct++) {
    int j = ct * 16 + ln;
#pragma unroll
    for (int r = 0; r < 8; r++) {
      int i = i0 + w * 16 + r + 8 * g;
      sc[ct][r] += bp[(size_t)i * NN + j];
    }
  }
#pragma unroll
  for (int r = 0; r < 8; r++) {
    float m = -INFINITY;
#pragma unroll
    for (int ct = 0; ct < 16; ct++) m = fmaxf(m, sc[ct][r]);
    for (int msk = 1; msk < 16; msk <<= 1) m = fmaxf(m, __shfl_xor(m, msk));
    float s = 0.f;
#pragma unroll
    for (int ct = 0; ct < 16; ct++) { float e = __expf(sc[ct][r] - m); sc[ct][r] = e; s += e; }
    for (int msk = 1; msk < 16; msk <<= 1) s += __shfl_xor(s, msk);
    float inv = 1.0f / s;
#pragma unroll
    for (int ct = 0; ct < 16; ct++) sc[ct][r] *= inv;
  }
  __syncthreads();

  const float* fw[3] = { f1w, f2w, f3w };
  for (int c = 0; c < 3; c++) {
#pragma unroll
    for (int ct = 0; ct < 16; ct++) {
      int j = ct * 16 + ln;
#pragma unroll
      for (int r = 0; r < 8; r++) {
        int ml = w * 16 + r + 8 * g;
        int i = i0 + ml;
        float ddx = psx[j] - psx[i];
        float ddy = psy[j] - psy[i];
        float ddz = psz[j] - psz[i];
        float dist = sqrtf(fmaxf(ddx * ddx + ddy * ddy + ddz * ddz, 1e-24f));
        float dc = ((c == 0) ? ddx : (c == 1) ? ddy : ddz) / (dist + 1e-5f);
        P[ml * 264 + j] = f2bf(sc[ct][r] * dc);
      }
    }
    __syncthreads();
    v8f ov[2]; ov[0] = zero8(); ov[1] = zero8();
#pragma unroll
    for (int ks = 0; ks < 8; ks++) {
      int k0 = ks * 32;
      FragB pf;
      const unsigned short* pp = P + (w * 16 + ln) * 264 + k0;
      pf.u[0] = *(const uint4*)(pp + 16 * g);
      pf.u[1] = *(const uint4*)(pp + 16 * g + 8);
#pragma unroll
      for (int ct = 0; ct < 2; ct++) {
        FragB vf;
        const unsigned short* vp = vtb + ((size_t)bh * 32 + ct * 16 + ln) * NN + k0 + 16 * g;
        vf.u[0] = *(const uint4*)vp;
        vf.u[1] = *(const uint4*)(vp + 8);
        ov[ct] = WMMA_BF16(pf.v, vf.v, ov[ct]);
      }
    }
    const float* wc = fw[c];
#pragma unroll
    for (int r = 0; r < 8; r++) {
      float s = ov[0][r] * wc[hh * HD + ln];
      int d1 = 16 + ln;
      if (d1 < HD) s += ov[1][r] * wc[hh * HD + d1];
      for (int msk = 1; msk < 16; msk <<= 1) s += __shfl_xor(s, msk);
      if (ln == 0) {
        int i = i0 + w * 16 + r + 8 * g;
        atomicAdd(outp + 4 + ((size_t)(b * NN + i)) * 3 + c, s);
      }
    }
    __syncthreads();
  }
}

// ================= energy head reduce + omask =================
__global__ void __launch_bounds__(256) energy_k(
    const float* __restrict__ tg, const float* __restrict__ en_w2, const float* __restrict__ en_b2,
    const float* __restrict__ eagg, const int* __restrict__ tags,
    const unsigned char* __restrict__ real_mask, float* __restrict__ outp)
{
  __shared__ float red[256];
  int row = blockIdx.x, t = threadIdx.x;
  const float* xr = tg + (size_t)row * DD;
  red[t] = xr[t] * en_w2[t] + xr[t + 256] * en_w2[t + 256] + xr[t + 512] * en_w2[t + 512];
  __syncthreads();
  for (int o = 128; o > 0; o >>= 1) { if (t < o) red[t] += red[t + o]; __syncthreads(); }
  if (t == 0) {
    float eng = red[0] + en_b2[0];
    int tag = tags[row];
    bool m = (tag > 0) && (real_mask[row] != 0);
    float e = eng * eagg[tag];
    int b = row >> 8;
    atomicAdd(outp + b, m ? e : 0.0f);
    outp[4 + BB * NN * 3 + row] = m ? 1.0f : 0.0f;
  }
}

__global__ void init_out_k(float* __restrict__ outp,
                           const float* __restrict__ f1b, const float* __restrict__ f2b,
                           const float* __restrict__ f3b)
{
  int idx = blockIdx.x * 256 + threadIdx.x;
  if (idx < 4) outp[idx] = 0.f;
  else if (idx < 4 + BB * NN * 3) {
    int c = (idx - 4) % 3;
    outp[idx] = (c == 0) ? f1b[0] : (c == 1) ? f2b[0] : f3b[0];
  }
}

// ================= host launcher =================
extern "C" void kernel_launch(void* const* d_in, const int* in_sizes, int n_in,
                              void* d_out, int out_size, void* d_ws, size_t ws_size,
                              hipStream_t stream)
{
  (void)in_sizes; (void)n_in; (void)out_size; (void)ws_size;
  const int*   atoms   = (const int*)d_in[0];
  const int*   tags    = (const int*)d_in[1];
  const float* pos     = (const float*)d_in[2];
  const unsigned char* rmask = (const unsigned char*)d_in[3];
  const float* atom_emb = (const float*)d_in[4];
  const float* tag_emb  = (const float*)d_in[5];
  const float* gbf_means = (const float*)d_in[6];
  const float* gbf_stds  = (const float*)d_in[7];
  const float* gbf_mul   = (const float*)d_in[8];
  const float* gbf_bias  = (const float*)d_in[9];
  const float* bp_w1 = (const float*)d_in[10];
  const float* bp_b1 = (const float*)d_in[11];
  const float* bp_w2 = (const float*)d_in[12];
  const float* bp_b2 = (const float*)d_in[13];
  const float* ep_w  = (const float*)d_in[14];
  const float* ep_b  = (const float*)d_in[15];
  const float* ln1_g = (const float*)d_in[16];
  const float* ln1_b = (const float*)d_in[17];
  const float* wqkv  = (const float*)d_in[18];
  const float* bqkv  = (const float*)d_in[19];
  const float* wo    = (const float*)d_in[20];
  const float* bo    = (const float*)d_in[21];
  const float* ln2_g = (const float*)d_in[22];
  const float* ln2_b = (const float*)d_in[23];
  const float* w1    = (const float*)d_in[24];
  const float* b1    = (const float*)d_in[25];
  const float* w2    = (const float*)d_in[26];
  const float* b2    = (const float*)d_in[27];
  const float* fln_g = (const float*)d_in[28];
  const float* fln_b = (const float*)d_in[29];
  const float* en_w1 = (const float*)d_in[30];
  const float* en_b1 = (const float*)d_in[31];
  const float* en_w2 = (const float*)d_in[32];
  const float* en_b2 = (const float*)d_in[33];
  const float* eagg  = (const float*)d_in[34];
  const float* nq_w  = (const float*)d_in[35];
  const float* nq_b  = (const float*)d_in[36];
  const float* nk_w  = (const float*)d_in[37];
  const float* nk_b  = (const float*)d_in[38];
  const float* nv_w  = (const float*)d_in[39];
  const float* nv_b  = (const float*)d_in[40];
  const float* f1_w  = (const float*)d_in[41];
  const float* f1_b  = (const float*)d_in[42];
  const float* f2_w  = (const float*)d_in[43];
  const float* f2_b  = (const float*)d_in[44];
  const float* f3_w  = (const float*)d_in[45];
  const float* f3_b  = (const float*)d_in[46];
  float* outp = (float*)d_out;

  // workspace carve-up
  char* ws = (char*)d_ws;
  size_t off = 0;
  auto alloc = [&](size_t bytes) -> char* {
    char* p = ws + off;
    off = (off + bytes + 255) & ~(size_t)255;
    return p;
  };
  unsigned short* gbf   = (unsigned short*)alloc((size_t)BB * NN * NN * KK * 2);
  float* biasbuf        = (float*)alloc((size_t)BB * HH * NN * NN * 4);
  float* edge_sum       = (float*)alloc((size_t)BB * NN * KK * 4);
  float* hbuf           = (float*)alloc((size_t)BB * NN * DD * 4);
  float* ybuf           = (float*)alloc((size_t)BB * NN * DD * 4);
  float* qkvbuf         = (float*)alloc((size_t)BB * NN * 3 * DD * 4);
  float* tbuf           = (float*)alloc((size_t)BB * NN * FF * 4);
  float* obuf           = (float*)alloc((size_t)BB * NN * DD * 4);
  unsigned short* qb    = (unsigned short*)alloc((size_t)BHN * NN * 32 * 2);
  unsigned short* kbuf  = (unsigned short*)alloc((size_t)BHN * NN * 32 * 2);
  unsigned short* vtb   = (unsigned short*)alloc((size_t)BHN * 32 * NN * 2);
  unsigned short* wqkvb = (unsigned short*)alloc((size_t)LL * 3 * DD * DD * 2);
  unsigned short* wob   = (unsigned short*)alloc((size_t)LL * DD * DD * 2);
  unsigned short* w1b   = (unsigned short*)alloc((size_t)LL * FF * DD * 2);
  unsigned short* w2b   = (unsigned short*)alloc((size_t)LL * DD * FF * 2);
  unsigned short* bpw1b = (unsigned short*)alloc((size_t)KK * KK * 2);
  unsigned short* bpw2b = (unsigned short*)alloc((size_t)HH * KK * 2);
  unsigned short* epwb  = (unsigned short*)alloc((size_t)DD * KK * 2);
  unsigned short* enw1b = (unsigned short*)alloc((size_t)DD * DD * 2);
  unsigned short* nqb   = (unsigned short*)alloc((size_t)DD * DD * 2);
  unsigned short* nkb   = (unsigned short*)alloc((size_t)DD * DD * 2);
  unsigned short* nvb   = (unsigned short*)alloc((size_t)DD * DD * 2);

  auto cvt = [&](const float* s, unsigned short* d, size_t n) {
    f2bf_k<<<(unsigned)((n + 255) / 256), 256, 0, stream>>>(s, d, (int)n);
  };
  cvt(wqkv, wqkvb, (size_t)LL * 3 * DD * DD);
  cvt(wo,   wob,   (size_t)LL * DD * DD);
  cvt(w1,   w1b,   (size_t)LL * FF * DD);
  cvt(w2,   w2b,   (size_t)LL * DD * FF);
  cvt(bp_w1, bpw1b, (size_t)KK * KK);
  cvt(bp_w2, bpw2b, (size_t)HH * KK);
  cvt(ep_w,  epwb,  (size_t)DD * KK);
  cvt(en_w1, enw1b, (size_t)DD * DD);
  cvt(nq_w, nqb, (size_t)DD * DD);
  cvt(nk_w, nkb, (size_t)DD * DD);
  cvt(nv_w, nvb, (size_t)DD * DD);

  const float scale = 0.20412414523193148f;  // 1/sqrt(24)
  const int M = BB * NN;                      // 1024

  gbf_k<<<BB * NN, 128, 0, stream>>>(atoms, pos, gbf_means, gbf_stds, gbf_mul, gbf_bias,
                                     gbf, edge_sum);
  bias_mlp_k<<<(BB * NN * NN) / 128, 256, 0, stream>>>(gbf, bpw1b, bp_b1, bpw2b, bp_b2,
                                                       atoms, biasbuf);
  node_emb_k<<<M, 256, 0, stream>>>(atoms, tags, atom_emb, tag_emb, hbuf);
  gemm_bf16_k<false, true><<<dim3(M / 128, DD / 128), 256, 0, stream>>>(
      edge_sum, KK, epwb, ep_b, hbuf, DD, hbuf, DD, KK);

  for (int l = 0; l < LL; l++) {
    ln_k<<<M, 256, 0, stream>>>(hbuf, ln1_g + l * DD, ln1_b + l * DD, ybuf);
    gemm_bf16_k<false, false><<<dim3(M / 128, (3 * DD) / 128), 256, 0, stream>>>(
        ybuf, DD, wqkvb + (size_t)l * 3 * DD * DD, bqkv + l * 3 * DD,
        nullptr, 0, qkvbuf, 3 * DD, DD);
    pack_qkv_k<<<BHN, 256, 0, stream>>>(qkvbuf, qb, kbuf, vtb, scale);
    attn_k<<<dim3(BHN, 4), 128, 0, stream>>>(qb, kbuf, vtb, biasbuf, obuf);
    gemm_bf16_k<false, true><<<dim3(M / 128, DD / 128), 256, 0, stream>>>(
        obuf, DD, wob + (size_t)l * DD * DD, bo + l * DD, hbuf, DD, hbuf, DD, DD);
    ln_k<<<M, 256, 0, stream>>>(hbuf, ln2_g + l * DD, ln2_b + l * DD, ybuf);
    gemm_bf16_k<true, false><<<dim3(M / 128, FF / 128), 256, 0, stream>>>(
        ybuf, DD, w1b + (size_t)l * FF * DD, b1 + l * FF, nullptr, 0, tbuf, FF, DD);
    gemm_bf16_k<false, true><<<dim3(M / 128, DD / 128), 256, 0, stream>>>(
        tbuf, FF, w2b + (size_t)l * DD * FF, b2 + l * DD, hbuf, DD, hbuf, DD, FF);
  }

  // final LN -> ybuf ("out" in reference)
  ln_k<<<M, 256, 0, stream>>>(hbuf, fln_g, fln_b, ybuf);

  // output init (energy zeros + node-head biases)
  init_out_k<<<(4 + BB * NN * 3 + BB * NN + 255) / 256, 256, 0, stream>>>(outp, f1_b, f2_b, f3_b);

  // energy head
  gemm_bf16_k<true, false><<<dim3(M / 128, DD / 128), 256, 0, stream>>>(
      ybuf, DD, enw1b, en_b1, nullptr, 0, tbuf, DD, DD);
  energy_k<<<M, 256, 0, stream>>>(tbuf, en_w2, en_b2, eagg, tags, rmask, outp);

  // node head: q/k/v projections packed into qkv layout
  gemm_bf16_k<false, false><<<dim3(M / 128, DD / 128), 256, 0, stream>>>(
      ybuf, DD, nqb, nq_b, nullptr, 0, qkvbuf, 3 * DD, DD);
  gemm_bf16_k<false, false><<<dim3(M / 128, DD / 128), 256, 0, stream>>>(
      ybuf, DD, nkb, nk_b, nullptr, 0, qkvbuf + DD, 3 * DD, DD);
  gemm_bf16_k<false, false><<<dim3(M / 128, DD / 128), 256, 0, stream>>>(
      ybuf, DD, nvb, nv_b, nullptr, 0, qkvbuf + 2 * DD, 3 * DD, DD);
  pack_qkv_k<<<BHN, 256, 0, stream>>>(qkvbuf, qb, kbuf, vtb, scale);
  attn_node_k<<<dim3(BHN, 4), 128, 0, stream>>>(qb, kbuf, vtb, biasbuf, pos,
                                                f1_w, f2_w, f3_w, outp);
}
